// ExpertGather_37117107372439
// MI455X (gfx1250) — matrix operand order repaired and checked by reading, not compiled, not run on hardware
//
#include <hip/hip_runtime.h>
#include <stdint.h>

typedef __attribute__((ext_vector_type(16))) __bf16 v16bf;
typedef __attribute__((ext_vector_type(8)))  __bf16 v8bf;
typedef __attribute__((ext_vector_type(4)))  __bf16 v4bf;
typedef __attribute__((ext_vector_type(8)))  float  v8f;
typedef __attribute__((ext_vector_type(4)))  int    v4i;

#define HEADS 16
#define KTOT  1024
#define CDIM  512
#define HD    64
#define NDIM  4096
#define ROWS  128   // K-rows per block
#define KC    64    // c-chunk staged per iteration
#define LDXF  68    // padded f32 LDS row stride for X (272B rows, 16B-aligned)
#define LDW   72    // padded bf16 LDS row stride for W^T

#if defined(__AMDGCN__) && __has_builtin(__builtin_amdgcn_global_load_async_to_lds_b128)
#define HAVE_ASYNC_LDS 1
#else
#define HAVE_ASYNC_LDS 0
#endif

__device__ __forceinline__ void wait_async0() {
#if HAVE_ASYNC_LDS
#if __has_builtin(__builtin_amdgcn_s_wait_asynccnt)
    __builtin_amdgcn_s_wait_asynccnt(0);
#else
    asm volatile("s_wait_asynccnt 0" ::: "memory");
#endif
#endif
}

// Copy 16 bytes global->LDS at byte offset OFF (added to BOTH addresses, per ISA).
// Builtin signature (from toolchain diagnostic): param0 = v4i addrspace(1)*,
// so cast through float* (drop const) then addrspacecast.
template <int OFF>
__device__ __forceinline__ void cp16_to_lds(const float* __restrict__ g, float* l) {
#if HAVE_ASYNC_LDS
    __builtin_amdgcn_global_load_async_to_lds_b128(
        (__attribute__((address_space(1))) v4i*)(float*)g,
        (__attribute__((address_space(3))) v4i*)l, OFF, 0);
#else
    *(float4*)((char*)l + OFF) = *(const float4*)((const char*)g + OFF);
#endif
}

// Build bf16 hi/lo split fragments from 16 f32 in LDS laid out per the
// 16-bit A-matrix 16x32 pattern: [p..p+7] and [p+16..p+23].
__device__ __forceinline__ void split_frag(const float* p, v16bf& hi, v16bf& lo) {
    float4 x0 = *(const float4*)(p);
    float4 x1 = *(const float4*)(p + 4);
    float4 x2 = *(const float4*)(p + 16);
    float4 x3 = *(const float4*)(p + 20);
    float f[16] = {x0.x, x0.y, x0.z, x0.w, x1.x, x1.y, x1.z, x1.w,
                   x2.x, x2.y, x2.z, x2.w, x3.x, x3.y, x3.z, x3.w};
    #pragma unroll
    for (int i = 0; i < 16; ++i) {
        const __bf16 h = (__bf16)f[i];
        hi[i] = h;
        lo[i] = (__bf16)(f[i] - (float)h);
    }
}

// out[b,h,k,j] = sum_i X[b, ind[b,h,k], i] * W[h,i,j]
// f32 realized as bf16x3 split-product on v_wmma_f32_16x16x32_bf16.
__global__ __launch_bounds__(256)
void expert_gather_wmma(const float* __restrict__ X,
                        const long long* __restrict__ ind,
                        const float* __restrict__ W,
                        float* __restrict__ out)
{
    __shared__ float  sXf[ROWS * LDXF];   // gathered X chunk, raw f32 (async DMA dest)
    __shared__ __bf16 sWhi[HD * LDW];     // W^T chunk, split bf16: [n][k]
    __shared__ __bf16 sWlo[HD * LDW];

    const int bid = blockIdx.x;
    const int kb  = bid & 7;                  // row-block within K
    const int h   = (bid >> 3) & (HEADS - 1);
    const int b   = bid >> 7;
    const int k0  = kb * ROWS;

    const int tid  = threadIdx.x;
    const int lane = tid & 31;
    const int wave = tid >> 5;                // 0..7, owns rows 16*wave..16*wave+15

    // ---- gather assignment: 2 threads per X row (32 floats each per chunk).
    //      NOTE: wave w stages exactly rows [16w,16w+16) == the rows it consumes,
    //      so s_wait_asynccnt 0 per-wave fully orders the async copies. ----
    const int grow  = tid >> 1;               // 0..127
    const int ghalf = tid & 1;
    const long long idx = ind[((size_t)b * HEADS + h) * KTOT + k0 + grow];
    const float* xrow = X + ((size_t)b * NDIM + (size_t)idx) * CDIM + ghalf * 32;
    float* xdst = sXf + grow * LDXF + ghalf * 32;

    // ---- W staging: thread owns column n, 4-consecutive-k groups (b64 stores) ----
    const int wn = tid & 63;                  // output column n
    const int wq = tid >> 6;                  // 0..3 -> k range [wq*16, wq*16+16)
    const float* wbase = W + (size_t)h * CDIM * HD + wn;

    v8f acc[4] = {v8f{}, v8f{}, v8f{}, v8f{}};

    const int am    = lane & 15;              // row (A) / col (B) within tile
    const int khalf = lane >> 4;              // lane half selects K sub-range

    for (int c0 = 0; c0 < CDIM; c0 += KC) {
        __syncthreads();   // previous chunk's compute done before restaging LDS

        // ---- stage gathered X rows: async DMA, no VGPR round-trip ----
        {
            const float* src = xrow + c0;
            cp16_to_lds<0>(src, xdst);
            cp16_to_lds<16>(src, xdst);
            cp16_to_lds<32>(src, xdst);
            cp16_to_lds<48>(src, xdst);
            cp16_to_lds<64>(src, xdst);
            cp16_to_lds<80>(src, xdst);
            cp16_to_lds<96>(src, xdst);
            cp16_to_lds<112>(src, xdst);
        }

        // ---- stage W chunk transposed: sW[n][k] = W[h][c0+k][n], split bf16 ----
        {
            #pragma unroll
            for (int p = 0; p < 4; ++p) {
                const int kbase = wq * 16 + p * 4;   // multiple of 4 -> 8B aligned
                v4bf hi, lo;
                #pragma unroll
                for (int q = 0; q < 4; ++q) {
                    const float wv = wbase[(size_t)(c0 + kbase + q) * HD];
                    const __bf16 whi = (__bf16)wv;
                    hi[q] = whi;
                    lo[q] = (__bf16)(wv - (float)whi);
                }
                *(v4bf*)(sWhi + wn * LDW + kbase) = hi;   // ds_store_b64
                *(v4bf*)(sWlo + wn * LDW + kbase) = lo;
            }
        }

        // ---- prefetch next chunk's X slice into L2/L0 (global_prefetch_b8) ----
        if (c0 + KC < CDIM) {
            __builtin_prefetch(xrow + c0 + KC, 0, 0);
            __builtin_prefetch(xrow + c0 + KC + 16, 0, 0);
        }

        __syncthreads();     // W staging visible to all waves
        wait_async0();       // this wave's X rows are resident in LDS

        // ---- compute: 2 K-steps of 32; split A on the fly (co-executes with WMMA) ----
        #pragma unroll
        for (int ks = 0; ks < 2; ++ks) {
            // A layout (16-bit A 16x32): lanes 0-15 -> K[0..7],[16..23]; lanes 16-31 -> +8
            const int akb = ks * 32 + khalf * 8;
            v16bf Ahi, Alo;
            split_frag(sXf + (16 * wave + am) * LDXF + akb, Ahi, Alo);

            // B layout (16-bit B 32x16): lanes 0-15 -> K[0..7],[16..23]; lanes 16-31 -> +8
            const int bkb = ks * 32 + khalf * 8;
            v16bf Bhi[4], Blo[4];
            #pragma unroll
            for (int t = 0; t < 4; ++t) {
                const __bf16* bh = sWhi + (16 * t + am) * LDW + bkb;
                const __bf16* bl = sWlo + (16 * t + am) * LDW + bkb;
                v8bf bh0 = *(const v8bf*)(bh);
                v8bf bh1 = *(const v8bf*)(bh + 16);
                v8bf bl0 = *(const v8bf*)(bl);
                v8bf bl1 = *(const v8bf*)(bl + 16);
                Bhi[t] = __builtin_shufflevector(bh0, bh1, 0,1,2,3,4,5,6,7,8,9,10,11,12,13,14,15);
                Blo[t] = __builtin_shufflevector(bl0, bl1, 0,1,2,3,4,5,6,7,8,9,10,11,12,13,14,15);
            }

            // 12 WMMAs: 4 independent accumulator chains for pipeline ILP
            #pragma unroll
            for (int t = 0; t < 4; ++t) {
                v8f a = acc[t];
                a = __builtin_amdgcn_wmma_f32_16x16x32_bf16(false, Alo, false, Bhi[t], (short)0, a, false, false);
                a = __builtin_amdgcn_wmma_f32_16x16x32_bf16(false, Ahi, false, Blo[t], (short)0, a, false, false);
                a = __builtin_amdgcn_wmma_f32_16x16x32_bf16(false, Ahi, false, Bhi[t], (short)0, a, false, false);
                acc[t] = a;
            }
        }
    }

    // ---- epilogue: C/D layout: VGPR j -> row j (lanes 0-15) / row j+8 (lanes 16-31),
    //      col = lane & 15.  NT stores: output never re-read, keep X resident in L2. ----
    const int orow0 = k0 + 16 * wave + 8 * khalf;
    const int ocol  = lane & 15;
    float* obase = out + (((size_t)b * HEADS + h) * KTOT + orow0) * HD + ocol;
    #pragma unroll
    for (int j = 0; j < 8; ++j) {
        __builtin_nontemporal_store(acc[0][j], obase + (size_t)j * HD + 0);
        __builtin_nontemporal_store(acc[1][j], obase + (size_t)j * HD + 16);
        __builtin_nontemporal_store(acc[2][j], obase + (size_t)j * HD + 32);
        __builtin_nontemporal_store(acc[3][j], obase + (size_t)j * HD + 48);
    }
}

extern "C" void kernel_launch(void* const* d_in, const int* in_sizes, int n_in,
                              void* d_out, int out_size, void* d_ws, size_t ws_size,
                              hipStream_t stream) {
    (void)in_sizes; (void)n_in; (void)out_size; (void)d_ws; (void)ws_size;
    const float*     X   = (const float*)d_in[0];
    const long long* ind = (const long long*)d_in[1];   // int64 indices
    const float*     W   = (const float*)d_in[2];
    float*           out = (float*)d_out;

    // B(8) * heads(16) * (K/ROWS = 8) row-blocks = 1024 workgroups, 8 waves each
    const int grid = 8 * HEADS * (KTOT / ROWS);
    expert_gather_wmma<<<grid, 256, 0, stream>>>(X, ind, W, out);
}
